// VectorQuantizer_21586505629900
// MI455X (gfx1250) — matrix operand order repaired and checked by reading, not compiled, not run on hardware
//
#include <hip/hip_runtime.h>

#define N_TOKENS  65536
#define EMBED_DIM 256
#define NUM_CODES 8192
#define NQ        (N_TOKENS * EMBED_DIM)

typedef __attribute__((ext_vector_type(16))) __bf16 v16bf;
typedef __attribute__((ext_vector_type(8)))  float  v8f;

// workspace layout (bytes)
#define WS_WH  0u                              // bf16 hi codebook: 8192*256*2 = 4 MB
#define WS_WL  (4u << 20)                      // bf16 lo codebook: 4 MB
#define WS_ES  (8u << 20)                      // |e_k|^2: 8192*4 = 32 KB
#define WS_IDX ((8u << 20) + 32768u)           // argmin indices: 65536*4 = 256 KB
#define WS_ACC ((8u << 20) + 32768u + 262144u) // double loss accumulator

// ---------------------------------------------------------------------------
// 1) Split codebook into bf16 hi/lo and compute |e_k|^2. One wave per code.
// ---------------------------------------------------------------------------
__global__ void vq_prep_weight(const float* __restrict__ w,
                               __bf16* __restrict__ wh, __bf16* __restrict__ wl,
                               float* __restrict__ es) {
  const int tid  = threadIdx.x;
  const int lane = tid & 31;
  const int code = blockIdx.x * 8 + (tid >> 5);
  const float* row = w + (size_t)code * EMBED_DIM;
  const int d0 = lane * 8;
  float s = 0.f;
#pragma unroll
  for (int i = 0; i < 8; ++i) {
    float v = row[d0 + i];
    __bf16 h = (__bf16)v;
    wh[(size_t)code * EMBED_DIM + d0 + i] = h;
    wl[(size_t)code * EMBED_DIM + d0 + i] = (__bf16)(v - (float)h);
    s += v * v;
  }
#pragma unroll
  for (int off = 16; off >= 1; off >>= 1) s += __shfl_down(s, off, 32);
  if (lane == 0) es[code] = s;
}

// ---------------------------------------------------------------------------
// 2) Fused GEMM + argmin. Grid: N/64 blocks x 128 threads (4 waves).
//    Each wave owns a 16-row tile, keeps A (hi/lo bf16 fragments) in VGPRs,
//    and streams all 512 16-code tiles with 3 WMMAs per 32-dim chunk (bf16x3).
//    Three independent accumulators (one per partial product) keep the WMMA
//    dependency chains at depth 8 instead of 24 so the XDL streams overlap.
// ---------------------------------------------------------------------------
__global__ void __launch_bounds__(128) vq_gemm_argmin(
    const float* __restrict__ x, const __bf16* __restrict__ wh,
    const __bf16* __restrict__ wl, const float* __restrict__ es,
    int* __restrict__ idx_out) {
  const int tid  = threadIdx.x;
  const int lane = tid & 31;
  const int wave = tid >> 5;
  const int m16  = lane & 15;   // A: row within tile; B/C: column within tile
  const int kg   = lane >> 4;   // k-group half
  const int rowbase = blockIdx.x * 64 + wave * 16;

  // Build A fragments in contiguous k order (same as B). Lanes 0-15 hold K=0..15,
  // lanes 16-31 hold K=16..31 of each 32-dim chunk (8 chunks for D=256).
  v16bf ah[8], al[8];
  const float* xr = x + (size_t)(rowbase + m16) * EMBED_DIM;
#pragma unroll
  for (int d = 0; d < 8; ++d) {
    const int k1 = 32 * d + 16 * kg;
#pragma unroll
    for (int i = 0; i < 8; ++i) {
      float v = xr[k1 + i];
      __bf16 h = (__bf16)v;
      ah[d][i] = h;
      al[d][i] = (__bf16)(v - (float)h);
    }
#pragma unroll
    for (int i = 0; i < 8; ++i) {
      float v = xr[k1 + 8 + i];
      __bf16 h = (__bf16)v;
      ah[d][8 + i] = h;
      al[d][8 + i] = (__bf16)(v - (float)h);
    }
  }

  float bestd[8];
  int   besti[8];
#pragma unroll
  for (int j = 0; j < 8; ++j) { bestd[j] = 3.0e38f; besti[j] = 0; }

  for (int nt = 0; nt < NUM_CODES / 16; ++nt) {
    const int n = nt * 16 + m16;  // this lane's column => code index
    // B in the same contiguous k order as A: lanes 0-15 hold K=0..15, lanes 16-31 hold K=16..31,
    // contiguous per lane => plain vector loads of the bf16 codebook rows.
    const __bf16* bhrow = wh + (size_t)n * EMBED_DIM + 16 * kg;
    const __bf16* blrow = wl + (size_t)n * EMBED_DIM + 16 * kg;
    __builtin_prefetch(bhrow + 16 * EMBED_DIM, 0, 1);  // next code tile
    __builtin_prefetch(blrow + 16 * EMBED_DIM, 0, 1);
    const float esn = es[n];

    // Three independent WMMA accumulation chains (depth 8 each).
    v8f c0 = {};  // xh . wh
    v8f c1 = {};  // xl . wh
    v8f c2 = {};  // xh . wl
#pragma unroll
    for (int d = 0; d < 8; ++d) {
      v16bf bh = *(const v16bf*)(bhrow + 32 * d);
      v16bf bl = *(const v16bf*)(blrow + 32 * d);
      c0 = __builtin_amdgcn_wmma_f32_16x16x32_bf16(false, ah[d], false, bh,
                                                   (short)0, c0, false, false);
      c1 = __builtin_amdgcn_wmma_f32_16x16x32_bf16(false, al[d], false, bh,
                                                   (short)0, c1, false, false);
      c2 = __builtin_amdgcn_wmma_f32_16x16x32_bf16(false, ah[d], false, bl,
                                                   (short)0, c2, false, false);
    }
    // distance discriminant: |e_n|^2 - 2 x.e_n  (|x|^2 constant per row)
#pragma unroll
    for (int j = 0; j < 8; ++j) {
      float dot  = c0[j] + c1[j] + c2[j];
      float dist = esn - 2.0f * dot;
      if (dist < bestd[j]) { bestd[j] = dist; besti[j] = n; }
    }
  }

  // Cross-lane argmin within each 16-lane half (C layout: VGPR j holds row
  // M=j for lanes 0-15 and M=j+8 for lanes 16-31). First-index tie-break.
#pragma unroll
  for (int j = 0; j < 8; ++j) {
    float d = bestd[j];
    int  bi = besti[j];
#pragma unroll
    for (int off = 1; off < 16; off <<= 1) {
      float od = __shfl_xor(d, off, 32);
      int   oi = __shfl_xor(bi, off, 32);
      if (od < d || (od == d && oi < bi)) { d = od; bi = oi; }
    }
    if (m16 == 0) idx_out[rowbase + j + 8 * kg] = bi;
  }
}

// ---------------------------------------------------------------------------
// 3) Gather quantized rows, emit quantized_st = x + (q - x) and indices,
//    accumulate sum((q - x)^2) into a double.
// ---------------------------------------------------------------------------
__global__ void vq_gather_loss(const float* __restrict__ x,
                               const float* __restrict__ w,
                               const int* __restrict__ idx,
                               float* __restrict__ out,
                               double* __restrict__ acc) {
  const int tid = threadIdx.x;
  const long long gid  = (long long)blockIdx.x * 256 + tid;
  const long long base = gid * 4;
  const int token = (int)(base >> 8);
  const int dim   = (int)(base & 255);
  const int k = idx[token];

  const float4 q  = *(const float4*)(w + (size_t)k * EMBED_DIM + dim);
  const float4 xi = *(const float4*)(x + base);
  float4 s, o;
  s.x = q.x - xi.x; s.y = q.y - xi.y; s.z = q.z - xi.z; s.w = q.w - xi.w;
  o.x = xi.x + s.x; o.y = xi.y + s.y; o.z = xi.z + s.z; o.w = xi.w + s.w;
  *(float4*)(out + base) = o;
  if (dim == 0) out[NQ + token] = (float)k;

  float local = s.x * s.x + s.y * s.y + s.z * s.z + s.w * s.w;
#pragma unroll
  for (int off = 16; off >= 1; off >>= 1) local += __shfl_down(local, off, 32);
  __shared__ float wsum[8];
  const int lane = tid & 31, wid = tid >> 5;
  if (lane == 0) wsum[wid] = local;
  __syncthreads();
  if (tid == 0) {
    float s8 = 0.f;
#pragma unroll
    for (int i = 0; i < 8; ++i) s8 += wsum[i];
    atomicAdd(acc, (double)s8);
  }
}

// ---------------------------------------------------------------------------
// 4) Finalize the three scalar losses.
// ---------------------------------------------------------------------------
__global__ void vq_finalize(const double* __restrict__ acc,
                            float* __restrict__ out) {
  const double l = *acc / (double)((long long)N_TOKENS * EMBED_DIM);
  const float lf = (float)l;
  out[NQ + N_TOKENS + 0] = lf + 0.25f * lf;  // vq_loss
  out[NQ + N_TOKENS + 1] = lf;               // codebook_loss
  out[NQ + N_TOKENS + 2] = lf;               // commit_loss
}

extern "C" void kernel_launch(void* const* d_in, const int* in_sizes, int n_in,
                              void* d_out, int out_size, void* d_ws,
                              size_t ws_size, hipStream_t stream) {
  const float* x = (const float*)d_in[0];  // (65536, 256)
  const float* w = (const float*)d_in[1];  // (8192, 256)
  float* out = (float*)d_out;
  char* ws = (char*)d_ws;

  __bf16* wh = (__bf16*)(ws + WS_WH);
  __bf16* wl = (__bf16*)(ws + WS_WL);
  float*  es = (float*)(ws + WS_ES);
  int*    idx = (int*)(ws + WS_IDX);
  double* acc = (double*)(ws + WS_ACC);

  hipMemsetAsync(acc, 0, sizeof(double), stream);
  vq_prep_weight<<<NUM_CODES / 8, 256, 0, stream>>>(w, wh, wl, es);
  vq_gemm_argmin<<<N_TOKENS / 64, 128, 0, stream>>>(x, wh, wl, es, idx);
  vq_gather_loss<<<NQ / 1024, 256, 0, stream>>>(x, w, idx, out, acc);
  vq_finalize<<<1, 1, 0, stream>>>(acc, out);
}
